// ImprovedOrthogonalProductQuantizer_33225867002070
// MI455X (gfx1250) — matrix operand order repaired and checked by reading, not compiled, not run on hardware
//
#include <hip/hip_runtime.h>
#include <stdint.h>

// Problem constants (from reference): B=8192, H=8, K=4096, D=1024, HD=128
#define BQ 8192
#define HQ 8
#define KQ 4096
#define HDQ 128

typedef __bf16        v16bf  __attribute__((ext_vector_type(16)));
typedef __bf16        bf16x4 __attribute__((ext_vector_type(4)));
typedef float         v8f    __attribute__((ext_vector_type(8)));
typedef float         f32x4  __attribute__((ext_vector_type(4)));
typedef unsigned int  u32x4  __attribute__((ext_vector_type(4)));

union Frag {
    v16bf bf;
    u32x4 q[2];
};

// ---------------------------------------------------------------------------
// Kernel 1: per-row (128-wide) L2 normalize fp32 -> bf16. One wave per row.
// ---------------------------------------------------------------------------
__global__ __launch_bounds__(256) void l2norm_rows_bf16(
    const float* __restrict__ in, __bf16* __restrict__ out, int rows)
{
    const int row  = blockIdx.x * 8 + (threadIdx.x >> 5);
    const int lane = threadIdx.x & 31;
    if (row >= rows) return;

    f32x4 v = *(const f32x4*)(in + (size_t)row * HDQ + lane * 4);
    float ss = v[0]*v[0] + v[1]*v[1] + v[2]*v[2] + v[3]*v[3];
#pragma unroll
    for (int m = 1; m < 32; m <<= 1) ss += __shfl_xor(ss, m, 32);
    const float inv = 1.0f / fmaxf(sqrtf(ss), 1e-12f);

    bf16x4 o;
#pragma unroll
    for (int i = 0; i < 4; ++i) o[i] = (__bf16)(v[i] * inv);
    *(bf16x4*)(out + (size_t)row * HDQ + lane * 4) = o;
}

// ---------------------------------------------------------------------------
// Kernel 2: per (b-tile=128, h): sims via bf16 WMMA, codebook chunks streamed
// into LDS with async DMA (double-buffered), distances NT-stored, fused
// argmax, raw-codebook gather for z_q.
// Block = 256 threads = 8 waves; wave w owns rows [m0, m0+16).
// ---------------------------------------------------------------------------
__global__ __launch_bounds__(256, 1) void pq_sims_argmax(
    const __bf16* __restrict__ zn,      // (B,H,HD) bf16, normalized
    const __bf16* __restrict__ cbn,     // (H,K,HD) bf16, normalized
    const float*  __restrict__ cb_raw,  // (H,K,HD) fp32, original codebooks
    float* __restrict__ zq,             // (B,D)
    float* __restrict__ oidx,           // (B,H) as float
    float* __restrict__ dist)           // (B,H,K)
{
    __shared__ __attribute__((aligned(16))) __bf16 As[128 * HDQ];     // 32 KB
    __shared__ __attribute__((aligned(16))) __bf16 Bs[2][128 * HDQ];  // 64 KB

    const int tid     = threadIdx.x;
    const int lane    = tid & 31;
    const int wave    = tid >> 5;
    const int b0      = blockIdx.x * 128;
    const int h       = blockIdx.y;
    const int m0      = wave * 16;
    const int halfsel = (lane >> 4) & 1;   // 0: lanes 0-15, 1: lanes 16-31
    const int l16     = lane & 15;

    // ---- Stage A tile: rows b0..b0+127 of head h (row stride H*HD in zn)
    {
        u32x4* dst = (u32x4*)As;
        for (int i = tid; i < 128 * 16; i += 256) {           // 16 quads/row
            const int row = i >> 4, j = i & 15;
            const u32x4* src =
                (const u32x4*)(zn + ((size_t)(b0 + row) * HQ + h) * HDQ);
            dst[(size_t)row * 16 + j] = src[j];
        }
    }

    // ---- Async DMA of one 128-entry codebook chunk (32 KB) into Bs[buf].
    // Each lane moves 8 x 16B via GLOBAL_LOAD_ASYNC_TO_LDS_B128 (ASYNCcnt).
    const char* cb_base =
        (const char*)(cbn + (size_t)h * KQ * HDQ);           // head base
    auto issue_chunk = [&](int chunk, int buf) {
        const char*    g = cb_base + (size_t)chunk * (128 * HDQ * 2);
        const uint32_t l = (uint32_t)(uintptr_t)&Bs[buf][0]; // LDS byte addr
#pragma unroll
        for (int r = 0; r < 8; ++r) {
            const int i = tid + r * 256;                     // quad index
            asm volatile("global_load_async_to_lds_b128 %0, %1, off"
                         :: "v"(l + i * 16),
                            "v"((const void*)(g + (size_t)i * 16))
                         : "memory");
        }
    };

    issue_chunk(0, 0);
    asm volatile("s_wait_asynccnt 0x0" ::: "memory");
    __syncthreads();

    // ---- A fragments (16x32 bf16 each), kept in VGPRs for the whole kernel.
    // ISA layout: lanes 0-15 row M=l16 hold K = kb..kb+7 and kb+16..kb+23
    // (kb = 32*s); lanes 16-31 hold K = kb+8..15 and kb+24..31.
    Frag a[4];
#pragma unroll
    for (int s = 0; s < 4; ++s) {
        const __bf16* p = &As[(m0 + l16) * HDQ + 32 * s + halfsel * 8];
        a[s].q[0] = *(const u32x4*)p;
        a[s].q[1] = *(const u32x4*)(p + 16);
    }

    // ---- Running per-lane argmax state. C-layout: VGPR v holds row
    // m0+v (lanes 0-15) / m0+8+v (lanes 16-31), column N = n0 + l16.
    float best[8];
    int   bidx[8];
#pragma unroll
    for (int v = 0; v < 8; ++v) { best[v] = -3.0e38f; bidx[v] = 0; }

    const size_t rstride = (size_t)HQ * KQ;                  // +1 batch row
    const size_t dbase =
        ((size_t)(b0 + m0 + halfsel * 8) * HQ + h) * (size_t)KQ + l16;

    const int nch = KQ / 128;
    for (int chunk = 0; chunk < nch; ++chunk) {
        const int buf = chunk & 1;
        if (chunk + 1 < nch) {
            issue_chunk(chunk + 1, buf ^ 1);                 // overlap DMA
            if (chunk + 2 < nch)                             // warm L2
                __builtin_prefetch(
                    cb_base + (size_t)(chunk + 2) * (128 * HDQ * 2) + tid * 128,
                    0, 1);
        }

#pragma unroll 2
        for (int sub = 0; sub < 8; ++sub) {
            const int n0 = chunk * 128 + sub * 16;

            // Load all 4 B fragments first (8x ds_load_b128 batched), then
            // issue the 4 WMMAs back-to-back.
            // B fragment (32x16): lanes 0-15 col N=l16 hold K=kb..kb+7 and
            // kb+16..kb+23, lanes 16-31 hold K=kb+8..kb+15 and
            // kb+24..kb+31.
            Frag bfr[4];
#pragma unroll
            for (int s = 0; s < 4; ++s) {
                const __bf16* p =
                    &Bs[buf][(sub * 16 + l16) * HDQ + 32 * s + halfsel * 8];
                bfr[s].q[0] = *(const u32x4*)p;
                bfr[s].q[1] = *(const u32x4*)(p + 16);
            }

            v8f acc = {};
#pragma unroll
            for (int s = 0; s < 4; ++s)
                acc = __builtin_amdgcn_wmma_f32_16x16x32_bf16(
                    false, a[s].bf, false, bfr[s].bf, (short)0, acc,
                    false, false);

            const size_t dp   = dbase + (size_t)n0;
            const int    ncol = n0 + l16;
#pragma unroll
            for (int v = 0; v < 8; ++v) {
                const float s = acc[v];
                __builtin_nontemporal_store(1.0f - s,
                                            &dist[dp + (size_t)v * rstride]);
                if (s > best[v]) { best[v] = s; bidx[v] = ncol; } // first-max
            }
        }

        asm volatile("s_wait_asynccnt 0x0" ::: "memory");    // next buf ready
        __syncthreads();
    }

    // ---- Cross-lane argmax within each 16-lane half (xor masks stay inside)
#pragma unroll
    for (int v = 0; v < 8; ++v) {
        float bv = best[v];
        int   bi = bidx[v];
#pragma unroll
        for (int m = 1; m < 16; m <<= 1) {
            const float ov = __shfl_xor(bv, m, 32);
            const int   oi = __shfl_xor(bi, m, 32);
            if (ov > bv || (ov == bv && oi < bi)) { bv = ov; bi = oi; }
        }
        best[v] = bv; bidx[v] = bi;
    }

    // ---- Indices output (rows m0+v from half 0, m0+8+v from half 1)
    if (l16 == 0) {
#pragma unroll
        for (int v = 0; v < 8; ++v) {
            const int row = b0 + m0 + halfsel * 8 + v;
            oidx[(size_t)row * HQ + h] = (float)bidx[v];
        }
    }

    // ---- z_q gather: whole wave copies each selected raw codebook row
#pragma unroll
    for (int rr = 0; rr < 16; ++rr) {
        const int src_lane = (rr < 8) ? 0 : 16;
        const int idx      = __shfl(bidx[rr & 7], src_lane, 32);
        const f32x4* srcp =
            (const f32x4*)(cb_raw + ((size_t)h * KQ + idx) * HDQ);
        f32x4* dstp =
            (f32x4*)(zq + ((size_t)(b0 + m0 + rr) * HQ + h) * HDQ);
        dstp[lane] = srcp[lane];
    }
}

// ---------------------------------------------------------------------------
extern "C" void kernel_launch(void* const* d_in, const int* in_sizes, int n_in,
                              void* d_out, int out_size, void* d_ws,
                              size_t ws_size, hipStream_t stream)
{
    const float* z  = (const float*)d_in[0];   // (B, D) fp32
    const float* cb = (const float*)d_in[1];   // (H, K, HD) fp32

    float* out  = (float*)d_out;
    float* zq   = out;                                   // B*D
    float* oidx = zq + (size_t)BQ * HQ * HDQ;            // B*H
    float* dist = oidx + (size_t)BQ * HQ;                // B*H*K

    __bf16* zn  = (__bf16*)d_ws;                         // B*H*HD bf16 (16MB)
    __bf16* cbn = zn + (size_t)BQ * HQ * HDQ;            // H*K*HD bf16 (8MB)

    l2norm_rows_bf16<<<(BQ * HQ) / 8, 256, 0, stream>>>(z, zn, BQ * HQ);
    l2norm_rows_bf16<<<(HQ * KQ) / 8, 256, 0, stream>>>(cb, cbn, HQ * KQ);
    pq_sims_argmax<<<dim3(BQ / 128, HQ), 256, 0, stream>>>(
        zn, cbn, cb, zq, oidx, dist);
}
